// ResLSTM_23338852287127
// MI455X (gfx1250) — matrix operand order repaired and checked by reading, not compiled, not run on hardware
//
#include <hip/hip_runtime.h>
#include <hip/hip_bf16.h>

// ---------------------------------------------------------------------------
// Fused ResLSTM for MI455X (gfx1250, wave32, WMMA).
// B=131072, N=62, T=5, H=32, C=3.  One wave = 16 batch rows (WMMA M dim).
// gates(16x128) = [x_t(16x64pad) | h(16x32)] x [W_ih^T ; W_hh^T] via
// v_wmma_f32_16x16x32_f16, 8 N-tiles x (2|1)+1 K-tiles per timestep.
// ---------------------------------------------------------------------------

typedef __attribute__((ext_vector_type(16))) _Float16 v16h;
typedef __attribute__((ext_vector_type(8)))  float    v8f;

#define WMMA16(A, Bm, Cm) \
  __builtin_amdgcn_wmma_f32_16x16x32_f16(false, (A), false, (Bm), (short)0, (Cm), false, false)

__device__ __forceinline__ float sigf(float x)  { return 1.0f / (1.0f + __expf(-x)); }
__device__ __forceinline__ float tanhq(float x) { return 2.0f / (1.0f + __expf(-2.0f * x)) - 1.0f; }
__device__ __forceinline__ float lrelu(float x) { return fmaxf(x, 0.01f * x); }

// LDS strides (halves), padded to avoid bank conflicts on strided gathers.
#define XROW 328   // x tile: [16][5*64 used, pad]
#define SROW 168   // h sequences: [16][5*32 used, pad]
#define HROW 40    // current h: [16][32 used, pad]

// Per-t-step LSTM layer: NKT = # of 32-wide K-tiles of the input sequence.
template <int NKT>
__device__ __forceinline__ void lstm_layer(
    const _Float16* __restrict__ inseq, int in_rowstride, int in_tstride,
    _Float16* __restrict__ hc, _Float16* __restrict__ outseq,
    const _Float16* __restrict__ WF, int fih, int fhh,
    const float* __restrict__ bih, const float* __restrict__ bhh, int lane)
{
  const int m    = lane & 15;
  const int off8 = (lane >> 4) * 8;

  // Per-lane fused biases (b_ih + b_hh) for the 2 column-tiles this lane owns.
  float bi[2], bf[2], bg[2], bo[2];
#pragma unroll
  for (int j = 0; j < 2; ++j) {
    int col = j * 16 + m;
    bi[j] = bih[col]      + bhh[col];
    bf[j] = bih[32 + col] + bhh[32 + col];
    bg[j] = bih[64 + col] + bhh[64 + col];
    bo[j] = bih[96 + col] + bhh[96 + col];
  }

  // h(t=-1) = 0
  for (int i = lane; i < 16 * HROW; i += 32) hc[i] = (_Float16)0.0f;

  float cst[2][8];
#pragma unroll
  for (int j = 0; j < 2; ++j)
#pragma unroll
    for (int r = 0; r < 8; ++r) cst[j][r] = 0.0f;

  for (int t = 0; t < 5; ++t) {
    // ---- gather A fragments (contiguous k order, matching the B fragments) ----
    v16h ax[2], ah;
#pragma unroll
    for (int e = 0; e < 16; ++e) {
      int kA = e + 2 * off8;   // K index inside 32-tile (contiguous: 16*half + e)
#pragma unroll
      for (int kt = 0; kt < NKT; ++kt)
        ax[kt][e] = inseq[m * in_rowstride + t * in_tstride + kt * 32 + kA];
      ah[e] = hc[m * HROW + kA];
    }

    // ---- 8 output tiles of 16x16 over the 128 gate columns ----
    v8f acc[8];
#pragma unroll
    for (int j = 0; j < 8; ++j) {
      v8f c = {};
      v16h bh = *(const v16h*)(WF + (size_t)(fhh + j) * 512 + lane * 16);
      c = WMMA16(ah, bh, c);
#pragma unroll
      for (int kt = 0; kt < NKT; ++kt) {
        v16h bb = *(const v16h*)(WF + (size_t)(fih + j * NKT + kt) * 512 + lane * 16);
        c = WMMA16(ax[kt], bb, c);
      }
      acc[j] = c;
    }

    // ---- gate nonlinearities; lane holds i/f/g/o for same (row, h) ----
#pragma unroll
    for (int j = 0; j < 2; ++j) {
#pragma unroll
      for (int r = 0; r < 8; ++r) {
        float gi = acc[j][r]     + bi[j];
        float gf = acc[j + 2][r] + bf[j];
        float gg = acc[j + 4][r] + bg[j];
        float go = acc[j + 6][r] + bo[j];
        float cc = sigf(gf) * cst[j][r] + sigf(gi) * tanhq(gg);
        cst[j][r] = cc;
        float h = sigf(go) * tanhq(cc);
        int row = r + off8, col = j * 16 + m;
        _Float16 hh = (_Float16)h;
        hc[row * HROW + col] = hh;
        outseq[row * SROW + t * 32 + col] = hh;
      }
    }
  }
}

// LayerNorm over 160 elems/row + LeakyReLU, written back in-place (f16).
__device__ __forceinline__ void ln_lrelu_inplace(
    _Float16* __restrict__ seq, const float* __restrict__ gw,
    const float* __restrict__ bw, int lane)
{
  int rown = lane & 15, half = lane >> 4;
  float sum = 0.0f, sq = 0.0f;
  for (int e = 0; e < 80; ++e) {
    float v = (float)seq[rown * SROW + half * 80 + e];
    sum += v; sq += v * v;
  }
  sum += __shfl_xor(sum, 16, 32);
  sq  += __shfl_xor(sq, 16, 32);
  float mu   = sum * (1.0f / 160.0f);
  float var  = sq * (1.0f / 160.0f) - mu * mu;
  float rstd = rsqrtf(var + 1e-5f);
  for (int e = 0; e < 80; ++e) {
    int idx = half * 80 + e;
    float v = (float)seq[rown * SROW + idx];
    float y = (v - mu) * rstd * gw[idx] + bw[idx];
    seq[rown * SROW + idx] = (_Float16)lrelu(y);
  }
}

__global__ void __launch_bounds__(128) reslstm_kernel(
    const float* __restrict__ data,
    const float* __restrict__ w_ih0, const float* __restrict__ w_hh0,
    const float* __restrict__ b_ih0, const float* __restrict__ b_hh0,
    const float* __restrict__ g0,    const float* __restrict__ be0,
    const float* __restrict__ w_ih1, const float* __restrict__ w_hh1,
    const float* __restrict__ b_ih1, const float* __restrict__ b_hh1,
    const float* __restrict__ g1,    const float* __restrict__ be1,
    const float* __restrict__ dw,    const float* __restrict__ db,
    float* __restrict__ out)
{
  // 40 weight fragments x 32 lanes x 16 halves, pre-swizzled B layout.
  __shared__ __attribute__((aligned(32))) _Float16 WF[40 * 512];
  __shared__ float CF[1152];   // g0,be0,g1,be1,dw,db staged as f32
  __shared__ __attribute__((aligned(32))) _Float16 XB[4][16 * XROW];
  __shared__ __attribute__((aligned(32))) _Float16 S0[4][16 * SROW];
  __shared__ __attribute__((aligned(32))) _Float16 S1[4][16 * SROW];
  __shared__ __attribute__((aligned(32))) _Float16 HC[4][16 * HROW];

  const int tid  = threadIdx.x;
  const int lane = tid & 31;
  const int wv   = tid >> 5;

  // ---- stage weight fragments (once per block, B-matrix 32x16 layout:
  //      k = e + 16*(lane>=16), col = 16*j + (lane&15)) ----
  for (int id = tid; id < 40 * 512; id += 128) {
    int f = id >> 9, r = id & 511, l = r >> 4, e = r & 15;
    int ln = l & 15, hi = (l >> 4) << 4;
    const float* W; int Kreal, col, k;
    if (f < 16)      { W = w_ih0; Kreal = 62; col = (f >> 1) * 16 + ln; k = (f & 1) * 32 + e + hi; }
    else if (f < 24) { W = w_hh0; Kreal = 32; col = (f - 16) * 16 + ln; k = e + hi; }
    else if (f < 32) { W = w_ih1; Kreal = 32; col = (f - 24) * 16 + ln; k = e + hi; }
    else             { W = w_hh1; Kreal = 32; col = (f - 32) * 16 + ln; k = e + hi; }
    float v = (k < Kreal) ? W[col * Kreal + k] : 0.0f;
    WF[id] = (_Float16)v;
  }
  for (int id = tid; id < 1123; id += 128) {
    float v;
    if      (id < 160)  v = g0[id];
    else if (id < 320)  v = be0[id - 160];
    else if (id < 480)  v = g1[id - 320];
    else if (id < 640)  v = be1[id - 480];
    else if (id < 1120) v = dw[id - 640];
    else                v = db[id - 1120];
    CF[id] = v;
  }
  __syncthreads();

  // ---- stage x tile: data is (B,1,N,T); x[b][t][n] = data[b][n*5+t].
  //      Fully coalesced global reads; scattered (cheap) LDS writes. ----
  const int bbase = (blockIdx.x * 4 + wv) * 16;
  const float* __restrict__ xg = data + (size_t)bbase * 310;
  _Float16* xb = XB[wv];
  for (int i = lane; i < 16 * 310; i += 32) {
    int row = i / 310, rem = i - row * 310;
    int n = rem / 5, t = rem - n * 5;
    xb[row * XROW + t * 64 + n] = (_Float16)xg[i];
  }
  for (int i = lane; i < 160; i += 32) {            // zero-pad n=62,63
    int row = i / 10, r2 = i - row * 10;
    xb[row * XROW + (r2 >> 1) * 64 + 62 + (r2 & 1)] = (_Float16)0.0f;
  }

  _Float16* s0 = S0[wv];
  _Float16* s1 = S1[wv];
  _Float16* hc = HC[wv];

  // ---- layer 0: input K-tiles = 2 (padded 64), frags ih=[0..15], hh=[16..23]
  lstm_layer<2>(xb, XROW, 64, hc, s0, WF, 0, 16, b_ih0, b_hh0, lane);
  ln_lrelu_inplace(s0, CF + 0, CF + 160, lane);

  // ---- layer 1: input K-tiles = 1, frags ih=[24..31], hh=[32..39]
  lstm_layer<1>(s0, SROW, 32, hc, s1, WF, 24, 32, b_ih1, b_hh1, lane);

  // ---- LN1 + residual + (160 -> 3) head, fused ----
  {
    int rown = lane & 15, half = lane >> 4;
    float sum = 0.0f, sq = 0.0f;
    for (int e = 0; e < 80; ++e) {
      float v = (float)s1[rown * SROW + half * 80 + e];
      sum += v; sq += v * v;
    }
    sum += __shfl_xor(sum, 16, 32);
    sq  += __shfl_xor(sq, 16, 32);
    float mu   = sum * (1.0f / 160.0f);
    float var  = sq * (1.0f / 160.0f) - mu * mu;
    float rstd = rsqrtf(var + 1e-5f);

    float a0 = 0.0f, a1 = 0.0f, a2 = 0.0f;
    for (int e = 0; e < 80; ++e) {
      int idx = half * 80 + e;
      float v = (float)s1[rown * SROW + idx];
      float y = lrelu((v - mu) * rstd * CF[320 + idx] + CF[480 + idx]);
      float s = y + (float)s0[rown * SROW + idx];     // residual
      a0 += s * CF[640 + idx];
      a1 += s * CF[800 + idx];
      a2 += s * CF[960 + idx];
    }
    a0 += __shfl_xor(a0, 16, 32);
    a1 += __shfl_xor(a1, 16, 32);
    a2 += __shfl_xor(a2, 16, 32);
    if (lane < 16) {
      float* o = out + (size_t)(bbase + rown) * 3;
      o[0] = lrelu(a0 + CF[1120]);
      o[1] = lrelu(a1 + CF[1121]);
      o[2] = lrelu(a2 + CF[1122]);
    }
  }
}

extern "C" void kernel_launch(void* const* d_in, const int* in_sizes, int n_in,
                              void* d_out, int out_size, void* d_ws, size_t ws_size,
                              hipStream_t stream) {
  (void)in_sizes; (void)n_in; (void)out_size; (void)d_ws; (void)ws_size;
  const float* data  = (const float*)d_in[0];
  const float* w_ih0 = (const float*)d_in[1];
  const float* w_hh0 = (const float*)d_in[2];
  const float* b_ih0 = (const float*)d_in[3];
  const float* b_hh0 = (const float*)d_in[4];
  const float* g0    = (const float*)d_in[5];
  const float* be0   = (const float*)d_in[6];
  const float* w_ih1 = (const float*)d_in[7];
  const float* w_hh1 = (const float*)d_in[8];
  const float* b_ih1 = (const float*)d_in[9];
  const float* b_hh1 = (const float*)d_in[10];
  const float* g1    = (const float*)d_in[11];
  const float* be1   = (const float*)d_in[12];
  const float* dw    = (const float*)d_in[13];
  const float* db    = (const float*)d_in[14];
  float* out = (float*)d_out;

  dim3 grid(131072 / 64);   // 64 batch rows per block (4 waves x 16)
  dim3 block(128);
  hipLaunchKernelGGL(reslstm_kernel, grid, block, 0, stream,
                     data, w_ih0, w_hh0, b_ih0, b_hh0, g0, be0,
                     w_ih1, w_hh1, b_ih1, b_hh1, g1, be1, dw, db, out);
}
